// AdaptiveGatedSLNNStep_89781996355837
// MI455X (gfx1250) — matrix operand order repaired and checked by reading, not compiled, not run on hardware
//
#include <hip/hip_runtime.h>

typedef __bf16 bf16;
typedef __attribute__((ext_vector_type(4)))  bf16  bf16x4;
typedef __attribute__((ext_vector_type(8)))  bf16  bf16x8;
typedef __attribute__((ext_vector_type(16))) bf16  bf16x16;
typedef __attribute__((ext_vector_type(8)))  float f32x8;

#define BATCH   4096
#define IN_DIM  1024
#define RDIM    2048
#define MAXR    4096
#define BM      128
#define BN      64
#define BK      32

// ---------------------------------------------------------------------------
// CDNA5 async global->LDS copy (GV mode, tracked by ASYNCcnt).
// Each active lane copies 16 bytes: LDS[vdst] = MEM[vaddr].
// ---------------------------------------------------------------------------
__device__ __forceinline__ void async_b128(bf16* lds_dst, const bf16* gsrc) {
  uint32_t loff = (uint32_t)(uintptr_t)(__attribute__((address_space(3))) bf16*)lds_dst;
  asm volatile("global_load_async_to_lds_b128 %0, %1, off"
               :: "v"(loff), "v"(gsrc) : "memory");
}

// Wait until ASYNCcnt <= N (async loads retire in order, so the oldest
// (current-tile) DMAs are complete once only the next tile's N remain).
template<int N>
__device__ __forceinline__ void wait_async_le() {
  asm volatile("s_wait_asynccnt %0" :: "n"(N) : "memory");
}

// Build a 16xbf16 WMMA fragment from two 16-byte LDS chunks (-> 2x ds_load_b128)
__device__ __forceinline__ bf16x16 make_frag(const bf16* p0, const bf16* p1) {
  bf16x8 lo = *reinterpret_cast<const bf16x8*>(p0);
  bf16x8 hi = *reinterpret_cast<const bf16x8*>(p1);
  return __builtin_shufflevector(lo, hi, 0,1,2,3,4,5,6,7,8,9,10,11,12,13,14,15);
}

__device__ __forceinline__ float sigmoidf_fast(float v) {
  return 1.0f / (1.0f + __expf(-v));
}

// ---------------------------------------------------------------------------
// Pass 0a: contiguous f32 -> bf16 convert (float4 granularity)
// ---------------------------------------------------------------------------
__global__ __launch_bounds__(256)
void cvt_contig_kernel(const float* __restrict__ src, bf16* __restrict__ dst, size_t n4)
{
  size_t idx = (size_t)blockIdx.x * blockDim.x + threadIdx.x;
  if (idx < n4) {
    float4 v = *reinterpret_cast<const float4*>(&src[idx * 4]);
    bf16x4 o;
    o.x = (bf16)v.x; o.y = (bf16)v.y; o.z = (bf16)v.z; o.w = (bf16)v.w;
    *reinterpret_cast<bf16x4*>(&dst[idx * 4]) = o;
  }
}

// ---------------------------------------------------------------------------
// Pass 0b: state[:, 0:2048] strided f32 -> bf16 convert
// ---------------------------------------------------------------------------
__global__ __launch_bounds__(256)
void cvt_state_kernel(const float* __restrict__ st, bf16* __restrict__ dst)
{
  const size_t ROW4 = RDIM / 4;
  size_t idx = (size_t)blockIdx.x * blockDim.x + threadIdx.x;
  size_t total = (size_t)BATCH * ROW4;
  if (idx < total) {
    size_t m  = idx / ROW4;
    size_t c4 = idx % ROW4;
    float4 v = *reinterpret_cast<const float4*>(&st[m * MAXR + c4 * 4]);
    bf16x4 o;
    o.x = (bf16)v.x; o.y = (bf16)v.y; o.z = (bf16)v.z; o.w = (bf16)v.w;
    *reinterpret_cast<bf16x4*>(&dst[m * RDIM + c4 * 4]) = o;
  }
}

// ---------------------------------------------------------------------------
// Pass 0c: Wr[0:2048, 0:2048] -> bf16, transposed to [n][k] (LDS-tiled)
// ---------------------------------------------------------------------------
__global__ __launch_bounds__(256)
void cvt_transpose_wr_kernel(const float* __restrict__ wr, bf16* __restrict__ dst)
{
  __shared__ bf16 tile[32][33];
  const int k0 = blockIdx.x * 32;
  const int n0 = blockIdx.y * 32;
  const int tx = threadIdx.x & 31;
  const int ty = threadIdx.x >> 5;
#pragma unroll
  for (int j = 0; j < 4; ++j) {
    int k = ty + j * 8;
    tile[tx][k] = (bf16)wr[(size_t)(k0 + k) * MAXR + n0 + tx];   // coalesced along n
  }
  __syncthreads();
#pragma unroll
  for (int j = 0; j < 4; ++j) {
    int n = ty + j * 8;
    dst[(size_t)(n0 + n) * RDIM + k0 + tx] = tile[n][tx];        // coalesced along k
  }
}

// ---------------------------------------------------------------------------
// Main fused kernel: 3 GEMMs (bf16 WMMA, f32 accum) + gated epilogue.
// Double-buffered LDS staging via async DMA (ASYNCcnt threshold waits).
// ---------------------------------------------------------------------------
__global__ __launch_bounds__(256)
void slnn_step_kernel(const bf16*  __restrict__ xb,    // [4096,1024]  bf16
                      const bf16*  __restrict__ stb,   // [4096,2048]  bf16
                      const float* __restrict__ st,    // [4096,4096]  f32 (epilogue prev)
                      const bf16*  __restrict__ wrtb,  // [2048,2048]  bf16, [n][k]
                      const bf16*  __restrict__ wib,   // [2048,1024]  bf16
                      const bf16*  __restrict__ wgb,   // [6144,1024]  bf16
                      float* __restrict__ out)         // [4096,4096]  f32
{
  __shared__ __align__(16) bf16 sA[2][BM * BK];        // ping-pong 128x32 A tiles
  __shared__ __align__(16) bf16 sB[2][4 * BN * BK];    // ping-pong B streams

  const int tid  = threadIdx.x;
  const int lane = tid & 31;
  const int wave = tid >> 5;
  const int lm   = lane & 15;
  const int lk   = lane >> 4;

  const int n0 = blockIdx.x * BN;
  const int m0 = blockIdx.y * BM;

  f32x8 acc_s[4] = {};
  f32x8 acc_i[4] = {};
  f32x8 acc_f[4] = {};
  f32x8 acc_o[4] = {};

  // chunk coordinates for async staging (16B = 8 bf16 per chunk)
  const int ar = tid >> 2, ac8 = tid & 3;

  // --- staging helpers (6 async instructions / wave for gates, 3 for reservoir) ---
  auto stage_gates = [&](int p, int kb) {
    bf16* A = sA[p];
    bf16* B = sB[p];
    async_b128(&A[ar * BK + ac8 * 8],        &xb[(size_t)(m0 + ar) * IN_DIM + kb + ac8 * 8]);
    async_b128(&A[(ar + 64) * BK + ac8 * 8], &xb[(size_t)(m0 + ar + 64) * IN_DIM + kb + ac8 * 8]);
    async_b128(&B[0 * BN * BK + ar * BK + ac8 * 8],
               &wib[(size_t)(n0 + ar) * IN_DIM + kb + ac8 * 8]);
#pragma unroll
    for (int s = 1; s < 4; ++s) {
      async_b128(&B[s * BN * BK + ar * BK + ac8 * 8],
                 &wgb[(size_t)(n0 + (s - 1) * RDIM + ar) * IN_DIM + kb + ac8 * 8]);
    }
  };
  auto stage_res = [&](int p, int kb) {
    bf16* A = sA[p];
    bf16* B = sB[p];
    async_b128(&A[ar * BK + ac8 * 8],        &stb[(size_t)(m0 + ar) * RDIM + kb + ac8 * 8]);
    async_b128(&A[(ar + 64) * BK + ac8 * 8], &stb[(size_t)(m0 + ar + 64) * RDIM + kb + ac8 * 8]);
    async_b128(&B[ar * BK + ac8 * 8],        &wrtb[(size_t)(n0 + ar) * RDIM + kb + ac8 * 8]);
  };

  // ------------- Pass 1: x @ {Wi, Wg_i, Wg_f, Wg_o}.T  (K = 1024) -------------
  stage_gates(0, 0);
  for (int kb = 0; kb < IN_DIM; kb += BK) {
    const int p = (kb >> 5) & 1;
    if (kb + BK < IN_DIM) {           // uniform branch: prefetch next tile
      stage_gates(p ^ 1, kb + BK);
      wait_async_le<6>();             // current tile's 6 DMAs complete
    } else {
      wait_async_le<0>();
    }
    __syncthreads();

    const bf16* aBase = &sA[p][(wave * 16 + lm) * BK];
    bf16x16 a = make_frag(aBase + lk * 16, aBase + lk * 16 + 8);

#pragma unroll
    for (int nt = 0; nt < 4; ++nt) {
      const int bOff = (nt * 16 + lm) * BK + lk * 16;
      const bf16* b0 = &sB[p][0 * (BN * BK) + bOff];
      const bf16* b1 = &sB[p][1 * (BN * BK) + bOff];
      const bf16* b2 = &sB[p][2 * (BN * BK) + bOff];
      const bf16* b3 = &sB[p][3 * (BN * BK) + bOff];
      bf16x16 bwi = make_frag(b0, b0 + 8);
      bf16x16 bi  = make_frag(b1, b1 + 8);
      bf16x16 bf  = make_frag(b2, b2 + 8);
      bf16x16 bo  = make_frag(b3, b3 + 8);
      acc_s[nt] = __builtin_amdgcn_wmma_f32_16x16x32_bf16(false, a, false, bwi, (short)0, acc_s[nt], false, false);
      acc_i[nt] = __builtin_amdgcn_wmma_f32_16x16x32_bf16(false, a, false, bi,  (short)0, acc_i[nt], false, false);
      acc_f[nt] = __builtin_amdgcn_wmma_f32_16x16x32_bf16(false, a, false, bf,  (short)0, acc_f[nt], false, false);
      acc_o[nt] = __builtin_amdgcn_wmma_f32_16x16x32_bf16(false, a, false, bo,  (short)0, acc_o[nt], false, false);
    }
    __syncthreads();                  // current buffer free for reuse
  }

  // ------------- Pass 2: prev_state @ Wr[:2048,:2048]  (K = 2048) -------------
  stage_res(0, 0);
  for (int kb = 0; kb < RDIM; kb += BK) {
    const int p = (kb >> 5) & 1;
    if (kb + BK < RDIM) {
      stage_res(p ^ 1, kb + BK);
      wait_async_le<3>();
    } else {
      wait_async_le<0>();
    }
    __syncthreads();

    const bf16* aBase = &sA[p][(wave * 16 + lm) * BK];
    bf16x16 a = make_frag(aBase + lk * 16, aBase + lk * 16 + 8);

#pragma unroll
    for (int nt = 0; nt < 4; ++nt) {
      const bf16* b = &sB[p][(nt * 16 + lm) * BK + lk * 16];
      bf16x16 br = make_frag(b, b + 8);
      acc_s[nt] = __builtin_amdgcn_wmma_f32_16x16x32_bf16(false, a, false, br, (short)0, acc_s[nt], false, false);
    }
    __syncthreads();
  }

  // ------------- Fused epilogue -------------
  // C layout: VGPR r -> M = r + 8*(lane>=16), N = lane%16
#pragma unroll
  for (int nt = 0; nt < 4; ++nt) {
    const int n = n0 + nt * 16 + lm;
#pragma unroll
    for (int r = 0; r < 8; ++r) {
      const int m = m0 + wave * 16 + r + lk * 8;
      float prev = st[(size_t)m * MAXR + n];
      float gi = sigmoidf_fast(acc_i[nt][r]);
      float gf = sigmoidf_fast(acc_f[nt][r]);
      float go = sigmoidf_fast(acc_o[nt][r]);
      float ns = 0.9f * (gf * prev) + 0.1f * tanhf(gi * acc_s[nt][r]);
      ns *= go;
      if (ns > 0.5f) ns -= 0.5f;
      out[(size_t)m * MAXR + n] = ns;
    }
  }
}

// Zero the padded columns [2048, 4096) of every row (d_out is poisoned by harness).
__global__ __launch_bounds__(256)
void zero_pad_kernel(float* __restrict__ out)
{
  const int PADW4 = (MAXR - RDIM) / 4;
  size_t idx = (size_t)blockIdx.x * blockDim.x + threadIdx.x;
  size_t total = (size_t)BATCH * PADW4;
  if (idx < total) {
    size_t m  = idx / PADW4;
    size_t c4 = idx % PADW4;
    float4 z = make_float4(0.f, 0.f, 0.f, 0.f);
    *reinterpret_cast<float4*>(&out[m * MAXR + RDIM + c4 * 4]) = z;
  }
}

extern "C" void kernel_launch(void* const* d_in, const int* in_sizes, int n_in,
                              void* d_out, int out_size, void* d_ws, size_t ws_size,
                              hipStream_t stream) {
  const float* x  = (const float*)d_in[0];   // inputs            [4096,1,1024]
  const float* st = (const float*)d_in[1];   // state             [4096,1,4096]
  const float* wr = (const float*)d_in[2];   // reservoir_weights [4096,4096]
  const float* wi = (const float*)d_in[3];   // input_weights     [4096,1024]
  const float* wg = (const float*)d_in[4];   // gate_weights      [12288,1024]
  float* out = (float*)d_out;

  // bf16 workspace layout (48 MB total)
  bf16* wsp  = (bf16*)d_ws;
  bf16* xb   = wsp;                                    // 4096*1024
  bf16* stb  = xb  + (size_t)BATCH * IN_DIM;           // 4096*2048
  bf16* wib  = stb + (size_t)BATCH * RDIM;             // 2048*1024
  bf16* wgb  = wib + (size_t)RDIM * IN_DIM;            // 6144*1024
  bf16* wrtb = wgb + (size_t)3 * RDIM * IN_DIM;        // 2048*2048 transposed

  // Pass 0: convert to bf16 (and transpose Wr) once
  {
    size_t n4;
    n4 = (size_t)BATCH * IN_DIM / 4;
    cvt_contig_kernel<<<(unsigned)((n4 + 255) / 256), 256, 0, stream>>>(x, xb, n4);
    size_t ns4 = (size_t)BATCH * (RDIM / 4);
    cvt_state_kernel<<<(unsigned)((ns4 + 255) / 256), 256, 0, stream>>>(st, stb);
    n4 = (size_t)RDIM * IN_DIM / 4;
    cvt_contig_kernel<<<(unsigned)((n4 + 255) / 256), 256, 0, stream>>>(wi, wib, n4);
    n4 = (size_t)3 * RDIM * IN_DIM / 4;
    cvt_contig_kernel<<<(unsigned)((n4 + 255) / 256), 256, 0, stream>>>(wg, wgb, n4);
    dim3 tg(RDIM / 32, RDIM / 32);
    cvt_transpose_wr_kernel<<<tg, 256, 0, stream>>>(wr, wrtb);
  }

  // Pass 1: fused GEMMs + epilogue
  dim3 grid(RDIM / BN, BATCH / BM);   // 32 x 32
  slnn_step_kernel<<<grid, 256, 0, stream>>>(xb, stb, st, wrtb, wib, wgb, out);

  // Pass 2: zero padded output region
  size_t pad4 = (size_t)BATCH * ((MAXR - RDIM) / 4);
  zero_pad_kernel<<<(unsigned)((pad4 + 255) / 256), 256, 0, stream>>>(out);
}
